// LSTM_64518998720972
// MI455X (gfx1250) — matrix operand order repaired and checked by reading, not compiled, not run on hardware
//
#include <hip/hip_runtime.h>
#include <cstddef>

// ---------------- problem constants ----------------
#define T_STEPS 512
#define BATCH   64
#define DIN     1024
#define HID     1024
#define G4H     4096     // 4*HID
#define NWG     64       // persistent workgroups (one per 16 hidden units)
#define NTHREADS 256     // 8 wave32 waves

// ---------------- WMMA types ----------------
typedef __attribute__((ext_vector_type(16))) __bf16 v16bf;
typedef __attribute__((ext_vector_type(8)))  float  v8f;

union Frag {
    uint4 q[2];   // 2 x 16B loads
    v16bf v;      // 16 bf16 operand for v_wmma_f32_16x16x32_bf16
};

__device__ __forceinline__ unsigned short f32_to_bf16(float f) {
    unsigned u = __float_as_uint(f);
    u += 0x7FFFu + ((u >> 16) & 1u);      // round-to-nearest-even
    return (unsigned short)(u >> 16);
}

// A fragment (16x32 bf16, row-major source, row stride 1024 elems):
// lane<16: K = base+{0..7, 16..23}; lane>=16: K = base+{8..15, 24..31}
__device__ __forceinline__ void load_frag_a(Frag& f, const unsigned short* p) {
    f.q[0] = *reinterpret_cast<const uint4*>(p);        // K +0..7
    f.q[1] = *reinterpret_cast<const uint4*>(p + 16);   // K +16..23
}
// B fragment (32x16 bf16) from pre-transposed W^T [N,K]: per lane K values
// of column n (lane<16: K {0..7, 16..23}, lane>=16: K {8..15, 24..31})
__device__ __forceinline__ void load_frag_b(Frag& f, const unsigned short* p) {
    f.q[0] = *reinterpret_cast<const uint4*>(p);
    f.q[1] = *reinterpret_cast<const uint4*>(p + 16);
}

__device__ __forceinline__ v8f wmma_bf16(const Frag& a, const Frag& b, v8f c) {
    return __builtin_amdgcn_wmma_f32_16x16x32_bf16(
        /*neg_a=*/false, a.v, /*neg_b=*/false, b.v,
        /*c_mod=*/(short)0, c, /*reuse_a=*/false, /*reuse_b=*/false);
}

__device__ __forceinline__ float sigmoidf_d(float x) {
    return 1.0f / (1.0f + __expf(-x));
}

// ---------------- device-wide split barrier ----------------
// sync[0] = arrival counter, sync[1] = generation
__device__ __forceinline__ void grid_barrier(unsigned* sync) {
    __syncthreads();
    if (threadIdx.x == 0) {
        __threadfence();
        unsigned g = atomicAdd(sync + 1, 0u);            // read generation
        if (atomicAdd(sync, 1u) == NWG - 1u) {
            atomicExch(sync, 0u);                        // reset counter
            __threadfence();
            atomicAdd(sync + 1, 1u);                     // release
        } else {
            while (*(volatile unsigned*)(sync + 1) == g)
                __builtin_amdgcn_s_sleep(2);
        }
        __threadfence();
    }
    __syncthreads();
}

// ---------------- helper kernels ----------------
__global__ void init_kernel(unsigned* sync, const float* __restrict__ h0,
                            float* __restrict__ out_head, int n) {
    int i = blockIdx.x * blockDim.x + threadIdx.x;
    if (i < 64) sync[i] = 0u;
    for (; i < n; i += gridDim.x * blockDim.x) out_head[i] = h0[i];  // out[0:2*B*H] = h0
}

__global__ void convert_bf16_kernel(const float* __restrict__ in,
                                    unsigned short* __restrict__ out, size_t n) {
    size_t i = (size_t)blockIdx.x * blockDim.x + threadIdx.x;
    size_t stride = (size_t)gridDim.x * blockDim.x;
    for (; i < n; i += stride) out[i] = f32_to_bf16(in[i]);
}

// W [K,N] fp32 row-major  ->  WT [N,K] bf16 row-major
__global__ void transpose_convert_kernel(const float* __restrict__ W,
                                         unsigned short* __restrict__ WT,
                                         int K, int N) {
    __shared__ float tile[32][33];
    int n0 = blockIdx.x * 32, k0 = blockIdx.y * 32;
    int tx = threadIdx.x, ty = threadIdx.y;     // block (32,8)
#pragma unroll
    for (int j = 0; j < 32; j += 8)
        tile[ty + j][tx] = W[(size_t)(k0 + ty + j) * N + (n0 + tx)];
    __syncthreads();
#pragma unroll
    for (int j = 0; j < 32; j += 8)
        WT[(size_t)(n0 + ty + j) * K + (k0 + tx)] = f32_to_bf16(tile[tx][ty + j]);
}

// ---------------- persistent LSTM layer kernel ----------------
// WG wg owns hidden units [wg*16, wg*16+16); computes all 4 gate tiles for
// them (N bases g*HID + wg*16), keeps c/h fp32 slices in LDS, publishes h in
// bf16 to a ping-pong global buffer with one grid barrier per timestep.
__global__ void __launch_bounds__(NTHREADS, 1)
lstm_layer_kernel(const unsigned short* __restrict__ xin,  // [T,B,1024] bf16
                  const unsigned short* __restrict__ WTi,  // [4H,1024] bf16 (W_ih^T)
                  const unsigned short* __restrict__ WTh,  // [4H,1024] bf16 (W_hh^T)
                  const float* __restrict__ bias,          // [4H]
                  const float* __restrict__ h0,            // [B,H] layer slice
                  const float* __restrict__ c0,            // [B,H]
                  const int* __restrict__ length,          // [B]
                  unsigned short* __restrict__ hping,      // [2][B*H] bf16
                  unsigned short* __restrict__ outb,       // [T,B,H] bf16 or null
                  float* __restrict__ outf,                // [T,B,H] f32 or null
                  unsigned* __restrict__ sync) {
    __shared__ float sG[4][BATCH][16];   // gate accumulators f,i,o,g
    __shared__ float sH[BATCH][16];      // fp32 carried hidden state slice
    __shared__ float sC[BATCH][16];      // fp32 carried cell state slice

    const int wg = blockIdx.x;           // 0..63
    const int tid = threadIdx.x;
    const int lane = tid & 31;
    const int wave = tid >> 5;           // 0..7
    const int mtile = wave & 3;          // batch-row tile (16 rows)
    const int gp = wave >> 2;            // 0 -> gates {f,i}, 1 -> gates {o,g}
    const int ug0 = wg * 16;             // first owned hidden unit

    // ---- init carried state; publish h0 (bf16) to ping buffer 0 ----
    for (int i = tid; i < BATCH * 16; i += NTHREADS) {
        int b = i >> 4, u = i & 15;
        float h = h0[(size_t)b * HID + ug0 + u];
        sH[b][u] = h;
        sC[b][u] = c0[(size_t)b * HID + ug0 + u];
        hping[(size_t)b * HID + ug0 + u] = f32_to_bf16(h);
    }

    // ---- warm this WG's weight slice into cache (global_prefetch_b8) ----
    {
        int r = tid & 127;               // 128 rows: 64 per matrix
        int half = tid >> 7;             // two threads per row
        int rr = r & 63;
        int gate = rr >> 4, u = rr & 15;
        const unsigned short* base =
            ((r >> 6) ? WTh : WTi) + (size_t)(gate * HID + ug0 + u) * 1024;
        const char* p = (const char*)base + half * 1024;
#pragma unroll
        for (int o = 0; o < 1024; o += 256) __builtin_prefetch(p + o, 0, 3);
    }
    grid_barrier(sync);

    // ---- per-wave fragment addressing (see layout notes above) ----
    const int arow = mtile * 16 + (lane & 15);
    const int akoff = (lane < 16) ? 0 : 8;
    const int ncol0 = (2 * gp + 0) * HID + ug0 + (lane & 15);
    const int ncol1 = (2 * gp + 1) * HID + ug0 + (lane & 15);
    const int bkoff = (lane < 16) ? 0 : 8;
    const unsigned short* wb0i = WTi + (size_t)ncol0 * 1024 + bkoff;
    const unsigned short* wb1i = WTi + (size_t)ncol1 * 1024 + bkoff;
    const unsigned short* wb0h = WTh + (size_t)ncol0 * 1024 + bkoff;
    const unsigned short* wb1h = WTh + (size_t)ncol1 * 1024 + bkoff;

    for (int t = 0; t < T_STEPS; ++t) {
        const unsigned short* xa =
            xin + (size_t)t * BATCH * DIN + (size_t)arow * DIN + akoff;
        const unsigned short* ha =
            hping + (size_t)(t & 1) * BATCH * HID + (size_t)arow * HID + akoff;

        v8f acc0 = {}, acc1 = {};
        Frag a, b0, b1;
        // gates += x_t @ W_ih   (K = 1024, 32 bf16 K-steps)
#pragma unroll 4
        for (int k = 0; k < 32; ++k) {
            load_frag_a(a, xa + k * 32);
            load_frag_b(b0, wb0i + k * 32);
            load_frag_b(b1, wb1i + k * 32);
            acc0 = wmma_bf16(a, b0, acc0);
            acc1 = wmma_bf16(a, b1, acc1);
        }
        // gates += h @ W_hh
#pragma unroll 4
        for (int k = 0; k < 32; ++k) {
            load_frag_a(a, ha + k * 32);
            load_frag_b(b0, wb0h + k * 32);
            load_frag_b(b1, wb1h + k * 32);
            acc0 = wmma_bf16(a, b0, acc0);
            acc1 = wmma_bf16(a, b1, acc1);
        }
        // scatter C tiles to LDS: VGPR e, lane l -> (M = e + 8*(l>=16), N = l&15)
        {
            int u = lane & 15;
            int mbase = mtile * 16 + ((lane >= 16) ? 8 : 0);
#pragma unroll
            for (int e = 0; e < 8; ++e) {
                sG[2 * gp + 0][mbase + e][u] = acc0[e];
                sG[2 * gp + 1][mbase + e][u] = acc1[e];
            }
        }
        __syncthreads();

        // ---- cell update (torch split order f,i,o,g) ----
        unsigned short* hn = hping + (size_t)((t + 1) & 1) * BATCH * HID;
        for (int i = tid; i < BATCH * 16; i += NTHREADS) {
            int b = i >> 4, u = i & 15;
            int ug = ug0 + u;
            float f  = sG[0][b][u] + bias[ug];
            float ii = sG[1][b][u] + bias[HID + ug];
            float o  = sG[2][b][u] + bias[2 * HID + ug];
            float g  = sG[3][b][u] + bias[3 * HID + ug];
            float cp = sC[b][u], hp = sH[b][u];
            float c1 = sigmoidf_d(f) * cp + sigmoidf_d(ii) * tanhf(g);
            float h1 = sigmoidf_d(o) * tanhf(c1);
            bool live = (t < length[b]);
            h1 = live ? h1 : hp;
            c1 = live ? c1 : cp;
            sH[b][u] = h1;
            sC[b][u] = c1;
            size_t oidx = (size_t)b * HID + ug;
            hn[oidx] = f32_to_bf16(h1);
            if (outb) outb[(size_t)t * BATCH * HID + oidx] = f32_to_bf16(h1);
            if (outf) outf[(size_t)t * BATCH * HID + oidx] = h1;
        }
        grid_barrier(sync);   // new h visible everywhere before next step
    }
}

// ---------------- host launch ----------------
extern "C" void kernel_launch(void* const* d_in, const int* in_sizes, int n_in,
                              void* d_out, int out_size, void* d_ws, size_t ws_size,
                              hipStream_t stream) {
    const float* x      = (const float*)d_in[0];
    const int*   length = (const int*)d_in[1];
    const float* h0     = (const float*)d_in[2];
    const float* c0     = (const float*)d_in[3];
    const float* W_ih0  = (const float*)d_in[4];
    const float* W_hh0  = (const float*)d_in[5];
    const float* b0     = (const float*)d_in[6];
    const float* W_ih1  = (const float*)d_in[7];
    const float* W_hh1  = (const float*)d_in[8];
    const float* b1     = (const float*)d_in[9];
    float* out = (float*)d_out;

    char* ws = (char*)d_ws;
    size_t off = 0;
    auto take = [&](size_t bytes) -> char* {
        char* p = ws + off;
        off += (bytes + 255) & ~(size_t)255;
        return p;
    };
    unsigned*       sync  = (unsigned*)take(256);
    unsigned short* xb    = (unsigned short*)take((size_t)T_STEPS * BATCH * DIN * 2);
    unsigned short* wti0  = (unsigned short*)take((size_t)G4H * DIN * 2);
    unsigned short* wth0  = (unsigned short*)take((size_t)G4H * HID * 2);
    unsigned short* wti1  = (unsigned short*)take((size_t)G4H * HID * 2);
    unsigned short* wth1  = (unsigned short*)take((size_t)G4H * HID * 2);
    unsigned short* out0b = (unsigned short*)take((size_t)T_STEPS * BATCH * HID * 2);
    unsigned short* hping = (unsigned short*)take((size_t)2 * BATCH * HID * 2);
    (void)ws_size; (void)in_sizes; (void)n_in; (void)out_size;

    // sync reset + out[0:2*B*H] = h0 (reference concatenates h0 in front)
    init_kernel<<<64, 256, 0, stream>>>(sync, h0, out, 2 * BATCH * HID);
    // x -> bf16
    convert_bf16_kernel<<<4096, 256, 0, stream>>>(x, xb, (size_t)T_STEPS * BATCH * DIN);
    // weights -> bf16, transposed to [N=4H, K]
    dim3 tb(32, 8), tg(G4H / 32, DIN / 32);
    transpose_convert_kernel<<<tg, tb, 0, stream>>>(W_ih0, wti0, DIN, G4H);
    transpose_convert_kernel<<<tg, tb, 0, stream>>>(W_hh0, wth0, HID, G4H);
    transpose_convert_kernel<<<tg, tb, 0, stream>>>(W_ih1, wti1, HID, G4H);
    transpose_convert_kernel<<<tg, tb, 0, stream>>>(W_hh1, wth1, HID, G4H);

    // layer 0: x -> out0 (bf16, feeds layer 1)
    lstm_layer_kernel<<<NWG, NTHREADS, 0, stream>>>(
        xb, wti0, wth0, b0, h0, c0, length, hping, out0b, nullptr, sync);
    // layer 1: out0 -> d_out[2*B*H ...] (fp32)
    lstm_layer_kernel<<<NWG, NTHREADS, 0, stream>>>(
        out0b, wti1, wth1, b1, h0 + (size_t)BATCH * HID, c0 + (size_t)BATCH * HID,
        length, hping, nullptr, out + (size_t)2 * BATCH * HID, sync);
}
